// GemmaSlidingLayer_38972533243958
// MI455X (gfx1250) — matrix operand order repaired and checked by reading, not compiled, not run on hardware
//
#include <hip/hip_runtime.h>
#include <math.h>

// ---------------- model constants ----------------
#define DM      2816          // D_MODEL
#define DDENSE  2112
#define NEXP    128
#define DFFN    704
#define TOPK    8
#define NH      16
#define DH      256
#define NKV     8
#define MAXCTX  4096
#define WIN     1024
#define WBASE   (MAXCTX - WIN)   // 3072
#define POS     (MAXCTX - 1)     // 4095
#define EPS     1e-6f

typedef __attribute__((ext_vector_type(2)))  float    v2f;
typedef __attribute__((ext_vector_type(4)))  float    v4f;
typedef __attribute__((ext_vector_type(8)))  float    v8f;
typedef __attribute__((ext_vector_type(16))) _Float16 v16h;
typedef __attribute__((ext_vector_type(4)))  unsigned v4u;
typedef __attribute__((ext_vector_type(4)))  int      v4i;
typedef __attribute__((ext_vector_type(8)))  int      v8i;

#ifndef __has_builtin
#define __has_builtin(x) 0
#endif
#if __has_builtin(__builtin_amdgcn_wmma_f32_16x16x4_f32)
#define USE_WMMA_F32 1
#else
#define USE_WMMA_F32 0
#endif
#if __has_builtin(__builtin_amdgcn_tensor_load_to_lds) && __has_builtin(__builtin_amdgcn_s_wait_tensorcnt)
#define HAVE_TDM 1
#else
#define HAVE_TDM 0
#endif

// TDM staging: each wave stages 16 rows x KT cols; LDS rows padded to KTP DWORDs
// (TDM pad: interval 32 DWORDs, amount 2 DWORDs) so the 16 row-parallel ds_load_b64
// hit distinct banks.
#define KT      32
#define KTP     (KT + 2)
#define STAGE_F (16 * KTP)        // floats per stage buffer (544)

// ---------------- helpers ----------------
__device__ inline float gelu_tanh(float x) {
    float x3 = x * x * x;
    float t  = tanhf(0.7978845608028654f * (x + 0.044715f * x3));
    return 0.5f * x * (1.0f + t);
}

__device__ inline unsigned lds_offset(const void* p) {
    // generic pointer to LDS: low 32 bits of the flat address are the LDS byte offset
    return (unsigned)(uintptr_t)p;
}

// block-wide sum for blockDim.x == 256 (8 waves). red must hold >= 9 floats.
__device__ inline float block_sum256(float v, float* red) {
    int lane = threadIdx.x & 31, wave = threadIdx.x >> 5;
#pragma unroll
    for (int m = 16; m > 0; m >>= 1) v += __shfl_xor(v, m, 32);
    if (lane == 0) red[wave] = v;
    __syncthreads();
    if (threadIdx.x == 0) {
        float s = 0.f;
        for (int i = 0; i < 8; ++i) s += red[i];
        red[8] = s;
    }
    __syncthreads();
    return red[8];
}

#if HAVE_TDM
// Issue one TDM descriptor: load a 16-row x KT-col f32 tile (row stride ldElems)
// from gsrc into LDS at lds_byte_off, with 2-DWORD padding after each 32-DWORD row.
// D# packing per CDNA5 ISA ch.8 (group0 128b, group1 256b; groups 2/3 unused -> 2D).
__device__ inline void tdm_issue(const float* gsrc, unsigned lds_byte_off, int ldElems) {
    unsigned long long ga = (unsigned long long)(uintptr_t)gsrc;
    v4u g0;
    g0[0] = 1u;                                             // count=1, user descriptor
    g0[1] = lds_byte_off;                                   // lds_addr
    g0[2] = (unsigned)(ga & 0xffffffffull);                 // global_addr[31:0]
    g0[3] = (unsigned)((ga >> 32) & 0x01ffffffull)          // global_addr[56:32]
          | (2u << 30);                                     // type = 2 ("image")
    v8i g1;
    g1[0] = (int)((2u << 16)                                // data_size = 4 bytes
                | (1u << 20)                                // pad_enable
                | (4u << 22)                                // pad_interval = 32 DWORDs
                | (1u << 25));                              // pad_amount  = 2 DWORDs
    g1[1] = (int)(((unsigned)KT & 0xffffu) << 16);          // tensor_dim0[15:0] (atomic_addr=0)
    g1[2] = (int)(16u << 16);                               // tensor_dim0[31:16]=0 | tensor_dim1[15:0]=16
    g1[3] = (int)(((unsigned)KT & 0xffffu) << 16);          // tensor_dim1[31:16]=0 | tile_dim0=KT
    g1[4] = 16;                                             // tile_dim1=16 rows, tile_dim2=0
    g1[5] = (int)(unsigned)ldElems;                         // tensor_dim0_stride[31:0]
    g1[6] = 0;                                              // stride0[47:32]=0, stride1[15:0]=0
    g1[7] = 0;
    v4i gz = {0, 0, 0, 0};
#if defined(__clang_major__) && (__clang_major__ >= 23)
    v8i gz8 = {0, 0, 0, 0, 0, 0, 0, 0};
    __builtin_amdgcn_tensor_load_to_lds(g0, g1, gz, gz, gz8, 0);
#else
    __builtin_amdgcn_tensor_load_to_lds(g0, g1, gz, gz, 0);
#endif
}
#endif // HAVE_TDM

// ---------------- WMMA GEMV core ----------------
// One wave computes y[m] = dot(Wtile[m,:], xs) for m = 0..15.
// Wtile: row-major, 16 rows of length D. xs: x in LDS. stage: this wave's
// 2*STAGE_F-float LDS staging area (double-buffered TDM destination).
// A 16x4 f32 layout (ISA 7.12.2): lanes 0-15 -> M=lane, K0/K1; lanes 16-31 -> M, K2/K3.
// B is x broadcast across all 16 N columns, so every column of D equals y.
__device__ inline v8f wave_gemv16(const float* __restrict__ Wtile,
                                  const float* __restrict__ xs, int D,
                                  float* __restrict__ stage) {
    const int  lane = threadIdx.x & 31;
    const int  m    = lane & 15;
    const bool hi   = lane >= 16;
    v8f acc = {0.f, 0.f, 0.f, 0.f, 0.f, 0.f, 0.f, 0.f};

#if USE_WMMA_F32 && HAVE_TDM
    // --- TDM double-buffered path: weights DMA'd to LDS, consumed via ds_load + WMMA ---
    const int koff = hi ? 2 : 0;
    const int nst  = D / KT;
    tdm_issue(Wtile, lds_offset(stage), D);
    for (int s = 0; s < nst; ++s) {
        const float* buf = stage + (s & 1) * STAGE_F;
        if (s + 1 < nst) {
            tdm_issue(Wtile + (size_t)(s + 1) * KT,
                      lds_offset(stage + ((s + 1) & 1) * STAGE_F), D);
            __builtin_amdgcn_s_wait_tensorcnt(1);   // oldest DMA (buffer s&1) done
        } else {
            __builtin_amdgcn_s_wait_tensorcnt(0);
        }
        const float* arow = buf + m * KTP + koff;   // padded row stride: no bank conflicts
        const float* xrow = xs + s * KT + koff;
#pragma unroll
        for (int kk = 0; kk < KT; kk += 4) {
            v2f a = { arow[kk], arow[kk + 1] };
            v2f b = { xrow[kk], xrow[kk + 1] };
            acc = __builtin_amdgcn_wmma_f32_16x16x4_f32(
                false, a, false, b, (short)0, acc, false, false);
        }
    }
#elif USE_WMMA_F32
    // --- direct-global f32 WMMA path ---
    const float* wrow = Wtile + (size_t)m * (size_t)D;
    const int koff = hi ? 2 : 0;
    for (int k = 0; k < D; k += 16) {
        __builtin_prefetch(wrow + k + 256, 0, 1);
#pragma unroll
        for (int kk = 0; kk < 16; kk += 4) {
            v2f a = { wrow[k + kk + koff], wrow[k + kk + koff + 1] };
            v2f b = { xs[k + kk + koff],   xs[k + kk + koff + 1]   };
            acc = __builtin_amdgcn_wmma_f32_16x16x4_f32(
                false, a, false, b, (short)0, acc, false, false);
        }
    }
#else
    // --- fallback: probe-confirmed f16 WMMA (K=32), f32 accumulate in C ---
    const float* wrow = Wtile + (size_t)m * (size_t)D;
    const int aoff = hi ? 16 : 0;   // A: lanes<16 K 0..15, lanes>=16 K 16..31
    const int boff = hi ? 16 : 0;   // B: lanes<16 K 0..15, lanes>=16 K 16..31
    for (int k = 0; k < D; k += 32) {
        __builtin_prefetch(wrow + k + 256, 0, 1);
        v16h A, B;
#pragma unroll
        for (int i = 0; i < 8; ++i)  A[i]     = (_Float16)wrow[k + aoff + i];
#pragma unroll
        for (int i = 0; i < 8; ++i)  A[8 + i] = (_Float16)wrow[k + aoff + 8 + i];
#pragma unroll
        for (int i = 0; i < 16; ++i) B[i]     = (_Float16)xs[k + boff + i];
        acc = __builtin_amdgcn_wmma_f32_16x16x32_f16(
            false, A, false, B, (short)0, acc, false, false);
    }
#endif
    return acc;
}

// Extract y[rowBase + 0..15] from the 16x16 C/D layout (N-replicated):
// lanes 0-15: VGPR v -> M=v (use lane 0); lanes 16-31: VGPR v -> M=v+8 (use lane 16).
__device__ inline void wave_store16(float* __restrict__ y, int rowBase,
                                    const v8f& acc, float scale, bool atomic) {
    const int lane = threadIdx.x & 31;
    if (lane == 0) {
#pragma unroll
        for (int i = 0; i < 8; ++i) {
            float v = acc[i] * scale;
            if (atomic) atomicAdd(y + rowBase + i, v);
            else        y[rowBase + i] = v;
        }
    } else if (lane == 16) {
#pragma unroll
        for (int i = 0; i < 8; ++i) {
            float v = acc[i] * scale;
            if (atomic) atomicAdd(y + rowBase + 8 + i, v);
            else        y[rowBase + 8 + i] = v;
        }
    }
}

// ---------------- generic GEMV kernel: y = W @ x, W is N x D row-major ----------------
__global__ void gemv_wmma_kernel(const float* __restrict__ W, const float* __restrict__ x,
                                 float* __restrict__ y, int N, int D) {
    __shared__ float stage[8][2 * STAGE_F];
    extern __shared__ float xs[];
    for (int i = threadIdx.x; i < D; i += blockDim.x) xs[i] = x[i];
    __syncthreads();
    int wave    = threadIdx.x >> 5;
    int rowBase = blockIdx.x * 128 + wave * 16;
    if (rowBase >= N) return;
    v8f acc = wave_gemv16(W + (size_t)rowBase * (size_t)D, xs, D, stage[wave]);
    wave_store16(y, rowBase, acc, 1.0f, false);
}

// ---------------- fused QKV GEMV (rows 0..4095 Wq, 4096..6143 Wk, 6144..8191 Wv) ----------------
__global__ void qkv_gemv_kernel(const float* __restrict__ Wq, const float* __restrict__ Wk,
                                const float* __restrict__ Wv, const float* __restrict__ x,
                                float* __restrict__ qkv) {
    __shared__ float stage[8][2 * STAGE_F];
    __shared__ float xs[DM];
    for (int i = threadIdx.x; i < DM; i += blockDim.x) xs[i] = x[i];
    __syncthreads();
    int wave    = threadIdx.x >> 5;
    int rowBase = blockIdx.x * 128 + wave * 16;
    const float* W; int r;
    if (rowBase < 4096)      { W = Wq; r = rowBase;        }
    else if (rowBase < 6144) { W = Wk; r = rowBase - 4096; }
    else                     { W = Wv; r = rowBase - 6144; }
    v8f acc = wave_gemv16(W + (size_t)r * DM, xs, DM, stage[wave]);
    wave_store16(qkv, rowBase, acc, 1.0f, false);
}

// ---------------- fused dense g/u GEMV (rows 0..2111 Wg, 2112..4223 Wu) ----------------
__global__ void densegu_gemv_kernel(const float* __restrict__ Wg, const float* __restrict__ Wu,
                                    const float* __restrict__ x, float* __restrict__ gu) {
    __shared__ float stage[8][2 * STAGE_F];
    __shared__ float xs[DM];
    for (int i = threadIdx.x; i < DM; i += blockDim.x) xs[i] = x[i];
    __syncthreads();
    int wave    = threadIdx.x >> 5;
    int rowBase = blockIdx.x * 128 + wave * 16;
    const float* W; int r;
    if (rowBase < DDENSE) { W = Wg; r = rowBase;          }
    else                  { W = Wu; r = rowBase - DDENSE; }
    v8f acc = wave_gemv16(W + (size_t)r * DM, xs, DM, stage[wave]);
    wave_store16(gu, rowBase, acc, 1.0f, false);
}

// ---------------- per-head RMSNorm (+RoPE for q/k) ----------------
// blocks: 0..15 q heads, 16..23 k heads, 24..31 v heads. 256 threads = DH.
__global__ void headnorm_rope_kernel(const float* __restrict__ qkv,
                                     const float* __restrict__ qw, const float* __restrict__ kw,
                                     const float* __restrict__ cosv, const float* __restrict__ sinv,
                                     float* __restrict__ qr, float* __restrict__ kr,
                                     float* __restrict__ vn) {
    __shared__ float sn[DH];
    __shared__ float red[16];
    int b = blockIdx.x, t = threadIdx.x;
    const float* in; const float* w; float* out; bool rope;
    if (b < 16)      { in = qkv + b * DH;                 w = qw;      out = qr + b * DH;        rope = true;  }
    else if (b < 24) { in = qkv + 4096 + (b - 16) * DH;   w = kw;      out = kr + (b - 16) * DH; rope = true;  }
    else             { in = qkv + 6144 + (b - 24) * DH;   w = nullptr; out = vn + (b - 24) * DH; rope = false; }
    float v   = in[t];
    float tot = block_sum256(v * v, red);
    float inv = rsqrtf(tot / (float)DH + EPS);
    float n   = v * inv * (w ? (1.0f + w[t]) : 1.0f);
    if (rope) {
        sn[t] = n;
        __syncthreads();
        float rot = (t < DH / 2) ? -sn[t + DH / 2] : sn[t - DH / 2];
        out[t] = n * cosv[t] + rot * sinv[t];
    } else {
        out[t] = n;
    }
}

// ---------------- KV cache copy + insert at POS (ext-vector float4, non-temporal) ----------------
__global__ void kvcopy_kernel(const v4f* __restrict__ kc, const v4f* __restrict__ vc,
                              const float* __restrict__ kr, const float* __restrict__ vn,
                              v4f* __restrict__ kout, v4f* __restrict__ vout) {
    size_t i  = (size_t)blockIdx.x * blockDim.x + threadIdx.x;  // 2,097,152 v4f per cache
    v4f k4 = __builtin_nontemporal_load(kc + i);
    v4f v4 = __builtin_nontemporal_load(vc + i);
    size_t e   = i * 4;
    int d      = (int)(e & (DH - 1));
    size_t row = e >> 8;                 // head*4096 + pos
    int pos    = (int)(row & (MAXCTX - 1));
    int head   = (int)(row >> 12);
    if (pos == POS) {
        const float* ks = kr + head * DH + d;
        const float* vs = vn + head * DH + d;
        k4.x += ks[0]; k4.y += ks[1]; k4.z += ks[2]; k4.w += ks[3];
        v4.x += vs[0]; v4.y += vs[1]; v4.z += vs[2]; v4.w += vs[3];
    }
    __builtin_nontemporal_store(k4, kout + i);
    __builtin_nontemporal_store(v4, vout + i);
}

// ---------------- windowed single-token attention: one block per head ----------------
__global__ void attn_kernel(const float* __restrict__ qr, const float* __restrict__ kout,
                            const float* __restrict__ vout, float* __restrict__ ctx) {
    __shared__ float qs[DH];
    __shared__ float sc[WIN];
    __shared__ float red[16];
    const int h = blockIdx.x, kvh = h >> 1;   // jnp.repeat(k,2): head h uses kv head h/2
    const int t = threadIdx.x, wave = t >> 5, lane = t & 31;
    qs[t] = qr[h * DH + t];
    __syncthreads();
    // scores: wave per position, lane-strided dot of length 256 (coalesced 128B chunks)
    const float* kbase = kout + ((size_t)kvh * MAXCTX + WBASE) * DH;
    for (int i = 0; i < WIN / 8; ++i) {
        int p = wave * (WIN / 8) + i;
        const float* krow = kbase + (size_t)p * DH;
        float s = 0.f;
#pragma unroll
        for (int j = 0; j < 8; ++j) s += qs[lane + 32 * j] * krow[lane + 32 * j];
#pragma unroll
        for (int m = 16; m > 0; m >>= 1) s += __shfl_xor(s, m, 32);
        if (lane == 0) sc[p] = s;
    }
    __syncthreads();
    // softmax over the 1024 window scores (mask is 0 inside window, -1e9 outside)
    float mx = -1e30f;
#pragma unroll
    for (int j = 0; j < 4; ++j) mx = fmaxf(mx, sc[t + 256 * j]);
#pragma unroll
    for (int m = 16; m > 0; m >>= 1) mx = fmaxf(mx, __shfl_xor(mx, m, 32));
    if (lane == 0) red[wave] = mx;
    __syncthreads();
    if (t == 0) { float v = red[0]; for (int i = 1; i < 8; ++i) v = fmaxf(v, red[i]); red[8] = v; }
    __syncthreads();
    mx = red[8];
    float ex[4], ls = 0.f;
#pragma unroll
    for (int j = 0; j < 4; ++j) { ex[j] = expf(sc[t + 256 * j] - mx); ls += ex[j]; }
#pragma unroll
    for (int m = 16; m > 0; m >>= 1) ls += __shfl_xor(ls, m, 32);
    if (lane == 0) red[wave] = ls;
    __syncthreads();
    if (t == 0) { float v = 0.f; for (int i = 0; i < 8; ++i) v += red[i]; red[9] = v; }
    __syncthreads();
    float inv = 1.0f / red[9];
#pragma unroll
    for (int j = 0; j < 4; ++j) sc[t + 256 * j] = ex[j] * inv;
    __syncthreads();
    // ctx[d] = sum_pos w[pos] * V[pos][d]  (coalesced across t=d)
    const float* vbase = vout + ((size_t)kvh * MAXCTX + WBASE) * DH;
    float acc = 0.f;
    for (int i = 0; i < WIN; ++i) acc += sc[i] * vbase[(size_t)i * DH + t];
    ctx[h * DH + t] = acc;
}

// ---------------- generic single-vector RMSNorm ----------------
__global__ void rmsnorm_kernel(const float* __restrict__ x, const float* __restrict__ w,
                               float* __restrict__ y, int D) {
    __shared__ float red[16];
    int t = threadIdx.x;
    float ss = 0.f;
    for (int i = t; i < D; i += 256) { float v = x[i]; ss += v * v; }
    float tot = block_sum256(ss, red);
    float inv = rsqrtf(tot / (float)D + EPS);
    for (int i = t; i < D; i += 256) y[i] = x[i] * inv * (1.0f + w[i]);
}

// ---------------- post-attention fused norms: x1, pre, hr, pre2 ----------------
__global__ void postattn_kernel(const float* __restrict__ a, const float* __restrict__ resid,
                                const float* __restrict__ w_post, const float* __restrict__ w_pre,
                                const float* __restrict__ rscale, const float* __restrict__ w_pre2,
                                float* __restrict__ x1, float* __restrict__ pre,
                                float* __restrict__ hr, float* __restrict__ pre2) {
    __shared__ float red[16];
    int t = threadIdx.x;
    float ss = 0.f;
    for (int i = t; i < DM; i += 256) { float v = a[i]; ss += v * v; }
    float inv1 = rsqrtf(block_sum256(ss, red) / (float)DM + EPS);
    float ss2 = 0.f;
    for (int i = t; i < DM; i += 256) {
        float xv = resid[i] + a[i] * inv1 * (1.0f + w_post[i]);
        x1[i] = xv;
        ss2 += xv * xv;
    }
    float inv2 = rsqrtf(block_sum256(ss2, red) / (float)DM + EPS);
    float rsc  = rsqrtf((float)DM);
    for (int i = t; i < DM; i += 256) {
        float xv = x1[i] * inv2;
        pre[i]  = xv * (1.0f + w_pre[i]);
        hr[i]   = xv * rscale[i] * rsc;
        pre2[i] = xv * (1.0f + w_pre2[i]);
    }
}

// ---------------- activations ----------------
__global__ void dense_act_kernel(const float* __restrict__ gu, float* __restrict__ hD) {
    int i = blockIdx.x * 256 + threadIdx.x;
    if (i < DDENSE) hD[i] = gelu_tanh(gu[i]) * gu[DDENSE + i];
}
__global__ void moe_act_kernel(const float* __restrict__ moegu, float* __restrict__ hh) {
    int i = blockIdx.x * 256 + threadIdx.x;   // i < 8*704 = 5632
    int s = i / DFFN, f = i % DFFN;
    hh[i] = gelu_tanh(moegu[s * 2 * DFFN + f]) * moegu[s * 2 * DFFN + DFFN + f];
}

// ---------------- router softmax + top-8 (also zeroes the MoE accumulator) ----------------
__global__ void router_topk_kernel(const float* __restrict__ logits, const float* __restrict__ pes,
                                   int* __restrict__ eidx, float* __restrict__ ew,
                                   float* __restrict__ moe_acc) {
    __shared__ float p[NEXP];
    __shared__ float sred[2];
    int t = threadIdx.x;                       // 128 threads
    for (int i = t; i < DM; i += NEXP) moe_acc[i] = 0.f;
    p[t] = logits[t];
    __syncthreads();
    if (t == 0) { float m = p[0]; for (int i = 1; i < NEXP; ++i) m = fmaxf(m, p[i]); sred[0] = m; }
    __syncthreads();
    float ex = expf(p[t] - sred[0]);
    p[t] = ex;
    __syncthreads();
    if (t == 0) { float s = 0.f; for (int i = 0; i < NEXP; ++i) s += p[i]; sred[1] = s; }
    __syncthreads();
    p[t] = p[t] / sred[1];
    __syncthreads();
    if (t == 0) {
        float tot = 0.f; int idx[TOPK]; float w[TOPK];
        for (int j = 0; j < TOPK; ++j) {
            int best = 0; float bv = p[0];
            for (int i = 1; i < NEXP; ++i) if (p[i] > bv) { bv = p[i]; best = i; }
            idx[j] = best; w[j] = bv; p[best] = -1.f; tot += bv;
        }
        for (int j = 0; j < TOPK; ++j) { eidx[j] = idx[j]; ew[j] = (w[j] / tot) * pes[idx[j]]; }
    }
}

// ---------------- sparse MoE: g/u GEMV for the 8 routed experts ----------------
// grid (11, 8): rows 0..703 -> Wg(e), 704..1407 -> Wu(e); expert rows live at e*704.
__global__ void moe_gu_kernel(const float* __restrict__ packs_Wg, const float* __restrict__ packs_Wu,
                              const float* __restrict__ x, const int* __restrict__ eidx,
                              float* __restrict__ out) {
    __shared__ float stage[8][2 * STAGE_F];
    __shared__ float xs[DM];
    for (int i = threadIdx.x; i < DM; i += blockDim.x) xs[i] = x[i];
    __syncthreads();
    int s = blockIdx.y, e = eidx[s];
    int wave = threadIdx.x >> 5;
    int row  = blockIdx.x * 128 + wave * 16;   // 0..1407
    const float* W; int r;
    if (row < DFFN) { W = packs_Wg; r = row;        }
    else            { W = packs_Wu; r = row - DFFN; }
    const float* Wtile = W + ((size_t)e * DFFN + (size_t)r) * (size_t)DM;
    v8f acc = wave_gemv16(Wtile, xs, DM, stage[wave]);
    wave_store16(out + s * 2 * DFFN, row, acc, 1.0f, false);
}

// ---------------- sparse MoE down: moe_acc += w_e * (Wd(e) @ hh_e), atomic f32 ----------------
// grid (22, 8). packs_Wd is (P, G*DM, DFFN): expert e rows start at e*DM.
__global__ void moe_down_kernel(const float* __restrict__ packs_Wd, const float* __restrict__ hh,
                                const int* __restrict__ eidx, const float* __restrict__ ew,
                                float* __restrict__ moe_acc) {
    __shared__ float stage[8][2 * STAGE_F];
    __shared__ float xs[DFFN];
    int s = blockIdx.y;
    for (int i = threadIdx.x; i < DFFN; i += blockDim.x) xs[i] = hh[s * DFFN + i];
    __syncthreads();
    int e = eidx[s];
    float wgt = ew[s];
    int wave = threadIdx.x >> 5;
    int row  = blockIdx.x * 128 + wave * 16;   // 0..2815
    const float* Wtile = packs_Wd + ((size_t)e * DM + (size_t)row) * (size_t)DFFN;
    v8f acc = wave_gemv16(Wtile, xs, DFFN, stage[wave]);
    wave_store16(moe_acc, row, acc, wgt, true);
}

// ---------------- final: h2 norm, (h1+h2) norm, residual + scale ----------------
__global__ void final_kernel(const float* __restrict__ moe, const float* __restrict__ w_post2,
                             const float* __restrict__ h1, const float* __restrict__ w_post,
                             const float* __restrict__ x1, const float* __restrict__ scal,
                             float* __restrict__ out) {
    __shared__ float red[16];
    int t = threadIdx.x;
    float ss = 0.f;
    for (int i = t; i < DM; i += 256) { float v = moe[i]; ss += v * v; }
    float inv_a = rsqrtf(block_sum256(ss, red) / (float)DM + EPS);
    float tv[11];
    float ss2 = 0.f;
#pragma unroll
    for (int j = 0; j < 11; ++j) {
        int i = t + 256 * j;
        float h2 = moe[i] * inv_a * (1.0f + w_post2[i]);
        float tt = h1[i] + h2;
        tv[j] = tt;
        ss2 += tt * tt;
    }
    float inv_b = rsqrtf(block_sum256(ss2, red) / (float)DM + EPS);
    float sc = scal[0];
#pragma unroll
    for (int j = 0; j < 11; ++j) {
        int i = t + 256 * j;
        float xf = tv[j] * inv_b * (1.0f + w_post[i]);
        out[i] = (x1[i] + xf) * sc;
    }
}

// ---------------- host side ----------------
extern "C" void kernel_launch(void* const* d_in, const int* in_sizes, int n_in,
                              void* d_out, int out_size, void* d_ws, size_t ws_size,
                              hipStream_t stream) {
    const float* x_in        = (const float*)d_in[0];
    const float* cosv        = (const float*)d_in[1];
    const float* sinv        = (const float*)d_in[2];
    const float* k_cache     = (const float*)d_in[3];
    const float* v_cache     = (const float*)d_in[4];
    const float* input_ln_w  = (const float*)d_in[7];
    const float* post_attn_w = (const float*)d_in[8];
    const float* pre_ffn_w   = (const float*)d_in[9];
    const float* post_ffn1_w = (const float*)d_in[10];
    const float* pre_ffn2_w  = (const float*)d_in[11];
    const float* post_ffn2_w = (const float*)d_in[12];
    const float* post_ffn_w  = (const float*)d_in[13];
    const float* Wq          = (const float*)d_in[14];
    const float* Wk          = (const float*)d_in[15];
    const float* Wv          = (const float*)d_in[16];
    const float* Wo          = (const float*)d_in[17];
    const float* q_norm_w    = (const float*)d_in[18];
    const float* k_norm_w    = (const float*)d_in[19];
    const float* dense_Wg    = (const float*)d_in[20];
    const float* dense_Wu    = (const float*)d_in[21];
    const float* dense_Wd    = (const float*)d_in[22];
    const float* router_Wp   = (const float*)d_in[23];
    const float* router_sc   = (const float*)d_in[24];
    const float* router_pes  = (const float*)d_in[25];
    const float* packs_Wg    = (const float*)d_in[26];
    const float* packs_Wu    = (const float*)d_in[27];
    const float* packs_Wd    = (const float*)d_in[28];
    const float* layer_scal  = (const float*)d_in[29];

    float* out  = (float*)d_out;                  // [0, 2816)
    float* kout = out + DM;                       // k_new: 8*4096*256
    float* vout = kout + (size_t)NKV * MAXCTX * DH;

    float* ws = (float*)d_ws;
    const size_t OFF_H     = 0;
    const size_t OFF_QKV   = OFF_H     + DM;        // 8192
    const size_t OFF_QR    = OFF_QKV   + 8192;      // 4096
    const size_t OFF_KR    = OFF_QR    + 4096;      // 2048
    const size_t OFF_VN    = OFF_KR    + 2048;      // 2048
    const size_t OFF_CTX   = OFF_VN    + 2048;      // 4096
    const size_t OFF_A     = OFF_CTX   + 4096;      // 2816
    const size_t OFF_X1    = OFF_A     + DM;
    const size_t OFF_PRE   = OFF_X1    + DM;
    const size_t OFF_HR    = OFF_PRE   + DM;
    const size_t OFF_PRE2  = OFF_HR    + DM;
    const size_t OFF_GU    = OFF_PRE2  + DM;        // 4224
    const size_t OFF_HD    = OFF_GU    + 2 * DDENSE;// 2112
    const size_t OFF_DB    = OFF_HD    + DDENSE;    // 2816
    const size_t OFF_H1    = OFF_DB    + DM;        // 2816
    const size_t OFF_LOG   = OFF_H1    + DM;        // 128
    const size_t OFF_EW    = OFF_LOG   + NEXP;      // 8
    const size_t OFF_EIDX  = OFF_EW    + TOPK;      // 8 ints
    const size_t OFF_MOEGU = OFF_EIDX  + TOPK;      // 8*1408
    const size_t OFF_HH    = OFF_MOEGU + 8 * 2 * DFFN; // 8*704
    const size_t OFF_MOE   = OFF_HH    + 8 * DFFN;  // 2816

    float* h     = ws + OFF_H;
    float* qkv   = ws + OFF_QKV;
    float* qr    = ws + OFF_QR;
    float* kr    = ws + OFF_KR;
    float* vn    = ws + OFF_VN;
    float* ctx   = ws + OFF_CTX;
    float* abuf  = ws + OFF_A;
    float* x1    = ws + OFF_X1;
    float* pre   = ws + OFF_PRE;
    float* hr    = ws + OFF_HR;
    float* pre2  = ws + OFF_PRE2;
    float* gu    = ws + OFF_GU;
    float* hD    = ws + OFF_HD;
    float* dbuf  = ws + OFF_DB;
    float* h1    = ws + OFF_H1;
    float* logit = ws + OFF_LOG;
    float* ew    = ws + OFF_EW;
    int*   eidx  = (int*)(ws + OFF_EIDX);
    float* moegu = ws + OFF_MOEGU;
    float* hh    = ws + OFF_HH;
    float* moe   = ws + OFF_MOE;

    // 1) h = rmsnorm(x, input_ln_w)
    rmsnorm_kernel<<<1, 256, 0, stream>>>(x_in, input_ln_w, h, DM);
    // 2) QKV projections (TDM-staged WMMA GEMV, 8192 rows)
    qkv_gemv_kernel<<<64, 256, 0, stream>>>(Wq, Wk, Wv, h, qkv);
    // 3) per-head norm + rope
    headnorm_rope_kernel<<<32, 256, 0, stream>>>(qkv, q_norm_w, k_norm_w, cosv, sinv, qr, kr, vn);
    // 4) KV cache copy + insert (2,097,152 float4 per cache, non-temporal)
    kvcopy_kernel<<<8192, 256, 0, stream>>>((const v4f*)k_cache, (const v4f*)v_cache,
                                            kr, vn, (v4f*)kout, (v4f*)vout);
    // 5) windowed attention
    attn_kernel<<<NH, 256, 0, stream>>>(qr, kout, vout, ctx);
    // 6) a = Wo @ ctx
    gemv_wmma_kernel<<<22, 256, NH * DH * sizeof(float), stream>>>(Wo, ctx, abuf, DM, NH * DH);
    // 7) x1 / pre / hr / pre2
    postattn_kernel<<<1, 256, 0, stream>>>(abuf, x_in, post_attn_w, pre_ffn_w,
                                           router_sc, pre_ffn2_w, x1, pre, hr, pre2);
    // 8) dense g/u
    densegu_gemv_kernel<<<33, 256, 0, stream>>>(dense_Wg, dense_Wu, pre, gu);
    // 9) hD = gelu(g) * u
    dense_act_kernel<<<9, 256, 0, stream>>>(gu, hD);
    // 10) dense down
    gemv_wmma_kernel<<<22, 256, DDENSE * sizeof(float), stream>>>(dense_Wd, hD, dbuf, DM, DDENSE);
    // 11) h1 = rmsnorm(dense, post_ffn_ln_1_w)
    rmsnorm_kernel<<<1, 256, 0, stream>>>(dbuf, post_ffn1_w, h1, DM);
    // 12) router logits (128 rows -> single block)
    gemv_wmma_kernel<<<1, 256, DM * sizeof(float), stream>>>(router_Wp, hr, logit, NEXP, DM);
    // 13) softmax + top-8 + zero moe accumulator
    router_topk_kernel<<<1, NEXP, 0, stream>>>(logit, router_pes, eidx, ew, moe);
    // 14) expert g/u for the 8 routed experts
    moe_gu_kernel<<<dim3(11, 8), 256, 0, stream>>>(packs_Wg, packs_Wu, pre2, eidx, moegu);
    // 15) hh = gelu(g) * u per expert
    moe_act_kernel<<<22, 256, 0, stream>>>(moegu, hh);
    // 16) expert down projections, weighted atomic accumulate
    moe_down_kernel<<<dim3(22, 8), 256, 0, stream>>>(packs_Wd, hh, eidx, ew, moe);
    // 17) final norms + residual + scale -> out[0..2816)
    final_kernel<<<1, 256, 0, stream>>>(moe, post_ffn2_w, h1, post_ffn_w, x1, layer_scal, out);
}
